// Cobrablock_8031588844262
// MI455X (gfx1250) — matrix operand order repaired and checked by reading, not compiled, not run on hardware
//
#include <hip/hip_runtime.h>
#include <stdint.h>

// ---------- types ----------
typedef __attribute__((ext_vector_type(16))) __bf16    v16bf;
typedef __attribute__((ext_vector_type(8)))  float     v8f;
typedef __attribute__((ext_vector_type(4)))  uint32_t  u32x4;

union Frag16 { v16bf v; u32x4 q[2]; uint32_t u[8]; };

static __device__ __forceinline__ uint16_t f2bf(float f) {
  uint32_t x = __float_as_uint(f);
  x += 0x7FFFu + ((x >> 16) & 1u);          // round-to-nearest-even
  return (uint16_t)(x >> 16);
}
static __device__ __forceinline__ uint32_t pack2(float lo, float hi) {
  return (uint32_t)f2bf(lo) | ((uint32_t)f2bf(hi) << 16);
}

#define WMMA_BF16(A, Bm, C) \
  __builtin_amdgcn_wmma_f32_16x16x32_bf16(false, (A), false, (Bm), (short)0, (C), false, false)

// ---------- weight pre-pack into WMMA B-fragment layout ----------
// Wp[((kt*(N/16)+nt)*32 + lane)*8 + i] = pack(W[kt*32+kh*16+2i][nt*16+(lane&15)],
//                                             W[kt*32+kh*16+2i+1][...]), kh = lane>>4
__global__ void pack_w_kernel(const float* __restrict__ W, uint32_t* __restrict__ Wp,
                              int K, int N) {
  int wid  = (blockIdx.x * blockDim.x + threadIdx.x) >> 5;
  int lane = threadIdx.x & 31;
  int ntiles = N >> 4;
  int tiles  = (K >> 5) * ntiles;
  if (wid >= tiles) return;
  int kt = wid / ntiles, nt = wid - kt * ntiles;
  int n     = nt * 16 + (lane & 15);
  int kbase = kt * 32 + ((lane >> 4) << 4);
  uint32_t* dst = Wp + ((size_t)wid * 32 + lane) * 8;
#pragma unroll
  for (int i = 0; i < 8; ++i) {
    int k = kbase + 2 * i;
    dst[i] = pack2(W[(size_t)k * N + n], W[(size_t)(k + 1) * N + n]);
  }
}

// ---------- layernorm -> bf16 ----------
__global__ void ln_bf16_kernel(const float* __restrict__ X, const float* __restrict__ g,
                               const float* __restrict__ bta, uint16_t* __restrict__ out) {
  __shared__ float s1[256], s2[256];
  int row = blockIdx.x, tid = threadIdx.x;
  const float* xr = X + (size_t)row * 1024;
  float sum = 0.f, sq = 0.f;
#pragma unroll
  for (int i = tid; i < 1024; i += 256) { float v = xr[i]; sum += v; sq += v * v; }
  s1[tid] = sum; s2[tid] = sq; __syncthreads();
  for (int st = 128; st > 0; st >>= 1) {
    if (tid < st) { s1[tid] += s1[tid + st]; s2[tid] += s2[tid + st]; }
    __syncthreads();
  }
  float mean = s1[0] * (1.0f / 1024.0f);
  float var  = s2[0] * (1.0f / 1024.0f) - mean * mean;
  float rs   = rsqrtf(var + 1e-5f);
  uint16_t* orow = out + (size_t)row * 1024;
#pragma unroll
  for (int i = tid; i < 1024; i += 256)
    orow[i] = f2bf((xr[i] - mean) * rs * g[i] + bta[i]);
}

// ---------- final: out = f + layernorm(f) ----------
__global__ void final_kernel(const float* __restrict__ F, const float* __restrict__ g,
                             const float* __restrict__ bta, float* __restrict__ out) {
  __shared__ float s1[256], s2[256];
  int row = blockIdx.x, tid = threadIdx.x;
  const float* fr = F + (size_t)row * 1024;
  float sum = 0.f, sq = 0.f;
#pragma unroll
  for (int i = tid; i < 1024; i += 256) { float v = fr[i]; sum += v; sq += v * v; }
  s1[tid] = sum; s2[tid] = sq; __syncthreads();
  for (int st = 128; st > 0; st >>= 1) {
    if (tid < st) { s1[tid] += s1[tid + st]; s2[tid] += s2[tid + st]; }
    __syncthreads();
  }
  float mean = s1[0] * (1.0f / 1024.0f);
  float var  = s2[0] * (1.0f / 1024.0f) - mean * mean;
  float rs   = rsqrtf(var + 1e-5f);
  float* orow = out + (size_t)row * 1024;
#pragma unroll
  for (int i = tid; i < 1024; i += 256) {
    float v = fr[i];
    orow[i] = v + (v - mean) * rs * g[i] + bta[i];
  }
}

// ---------- RoPE (f32 in, bf16 out), half-split rotation ----------
__global__ void rope_kernel(const float* __restrict__ X, uint16_t* __restrict__ out) {
  int row = blockIdx.x;
  int s = row & 2047;                       // S = 2048
  const float* xr = X + (size_t)row * 1024;
  uint16_t* orow = out + (size_t)row * 1024;
#pragma unroll
  for (int i = threadIdx.x; i < 512; i += 256) {
    float freq = __powf(10000.0f, -(float)i * (1.0f / 512.0f));
    float ang = (float)s * freq;
    float sn, cs; __sincosf(ang, &sn, &cs);
    float x1 = xr[i], x2 = xr[i + 512];
    orow[i]       = f2bf(x1 * cs - x2 * sn);
    orow[i + 512] = f2bf(x1 * sn + x2 * cs);
  }
}

// ---------- valh (B*S,256) -> vT (B,H,32,S) ----------
__global__ void vtrans_kernel(const uint16_t* __restrict__ V, uint16_t* __restrict__ VT) {
  int idx = blockIdx.x * blockDim.x + threadIdx.x;   // B*H*32*S = 1<<20
  int s = idx & 2047;
  int k = (idx >> 11) & 31;
  int h = (idx >> 16) & 7;
  int b = idx >> 19;
  VT[idx] = V[(((size_t)b * 2048 + s) * 256) + h * 32 + k];
}

// ---------- SiLU(g) * u -> bf16 ----------
__global__ void silu_mul_kernel(const float* __restrict__ G, const float* __restrict__ U,
                                uint16_t* __restrict__ out, int n) {
  int i = blockIdx.x * blockDim.x + threadIdx.x;
  if (i < n) {
    float g = G[i];
    float s = 1.0f / (1.0f + __expf(-g));
    out[i] = f2bf(g * s * U[i]);
  }
}

// ---------- bf16 WMMA GEMM with async-to-LDS double-buffered B staging ----------
// C[M,N] = A[M,K] * W + bias (+resid).  block = 256 thr (8 waves), tile 128Mx64N.
// B tile (32k x 64n bf16 = 4KB, already fragment-ordered) is staged per block via
// GLOBAL_LOAD_ASYNC_TO_LDS_B128 (ASYNCcnt), so the 8 waves share one copy.
__global__ void gemm_kernel(const uint16_t* __restrict__ A, const uint32_t* __restrict__ Wp,
                            const float* __restrict__ bias, const float* __restrict__ resid,
                            float* __restrict__ Cf, uint16_t* __restrict__ Cb,
                            int N, int K) {
  __shared__ __align__(16) uint32_t bstage[2][1024];   // 2 x 4KB
  int tid   = threadIdx.x;
  int lane  = tid & 31;
  int wave  = tid >> 5;
  int m0    = blockIdx.y * 128 + wave * 16;
  int n0    = blockIdx.x * 64;
  int ntiles = N >> 4;
  int khalf = lane >> 4;
  int l15   = lane & 15;
  const uint16_t* arow = A + (size_t)(m0 + l15) * K;

  // each thread owns one 16B slice of the 4KB tile copy
  const uint32_t* wbase = Wp + ((size_t)(n0 >> 4) * 32) * 8 + (size_t)tid * 4;
  size_t wstride = (size_t)ntiles * 256;               // dwords per kt step

  v8f z8 = {0.f,0.f,0.f,0.f,0.f,0.f,0.f,0.f};
  v8f acc[4] = {z8, z8, z8, z8};

  int ktt = K >> 5;
  {
    uint32_t loff = (uint32_t)(uintptr_t)(&bstage[0][tid * 4]);
    asm volatile("global_load_async_to_lds_b128 %0, %1, off"
                 :: "v"(loff), "v"(wbase) : "memory");
  }
  for (int kt = 0; kt < ktt; ++kt) {
    if (kt + 1 < ktt) {
      uint32_t loff = (uint32_t)(uintptr_t)(&bstage[(kt + 1) & 1][tid * 4]);
      const uint32_t* g = wbase + (size_t)(kt + 1) * wstride;
      asm volatile("global_load_async_to_lds_b128 %0, %1, off"
                   :: "v"(loff), "v"(g) : "memory");
      asm volatile("s_wait_asynccnt 0x1" ::: "memory");  // current tile landed
    } else {
      asm volatile("s_wait_asynccnt 0x0" ::: "memory");
    }
    __syncthreads();                                     // tile visible to all waves

    Frag16 a;
    a.q[0] = *(const u32x4*)(arow + kt * 32 + khalf * 16);
    a.q[1] = *(const u32x4*)(arow + kt * 32 + khalf * 16 + 8);
    const uint32_t* bst = &bstage[kt & 1][(size_t)lane * 8];
#pragma unroll
    for (int j = 0; j < 4; ++j) {
      Frag16 b;
      b.q[0] = *(const u32x4*)(bst + j * 256);
      b.q[1] = *(const u32x4*)(bst + j * 256 + 4);
      acc[j] = WMMA_BF16(a.v, b.v, acc[j]);
    }
    __syncthreads();                                     // done reading before overwrite
  }
#pragma unroll
  for (int j = 0; j < 4; ++j) {
    int ncol = n0 + j * 16 + l15;
    float bv = bias ? bias[ncol] : 0.0f;
#pragma unroll
    for (int r = 0; r < 8; ++r) {
      int m = m0 + r + khalf * 8;
      size_t idx = (size_t)m * N + ncol;
      float v = acc[j][r] + bv;
      if (resid) v += resid[idx];
      if (Cf) Cf[idx] = v;
      if (Cb) Cb[idx] = f2bf(v);
    }
  }
}

// ---------- flash attention: one wave per (b, h, 16-query tile) ----------
// QH,KH: (B*S, 256) bf16 ; VT: (B,H,32,S) bf16 ; AO: (B*S, 256) bf16
__global__ void attn_kernel(const uint16_t* __restrict__ QH, const uint16_t* __restrict__ KH,
                            const uint16_t* __restrict__ VT, uint16_t* __restrict__ AO) {
  __shared__ uint16_t pls[8][16 * 32];
  int lane  = threadIdx.x & 31;
  int wave  = threadIdx.x >> 5;
  int wid   = blockIdx.x * 8 + wave;         // 0..2047
  int q0    = (wid & 127) << 4;
  int h     = (wid >> 7) & 7;
  int b     = wid >> 10;
  int khalf = lane >> 4;
  int l15   = lane & 15;

  // qh A-fragment (16 queries x 32 head-dim) -- loop invariant
  const uint16_t* qrow = QH + ((size_t)(b * 2048 + q0 + l15)) * 256 + h * 32;
  Frag16 aq;
  aq.q[0] = *(const u32x4*)(qrow + khalf * 16);
  aq.q[1] = *(const u32x4*)(qrow + khalf * 16 + 8);

  const uint16_t* kb = KH + ((size_t)b * 2048) * 256 + h * 32;
  const uint16_t* vb = VT + (((size_t)(b * 8 + h)) * 32) * 2048;

  v8f z8 = {0.f,0.f,0.f,0.f,0.f,0.f,0.f,0.f};
  v8f acc0 = z8, acc1 = z8;                  // 16 x 32 output accumulator
  float mrow[8], lrow[8];
#pragma unroll
  for (int r = 0; r < 8; ++r) { mrow[r] = -1e30f; lrow[r] = 0.0f; }
  const float scale = 0.17677669529663687f;  // 1/sqrt(32)
  uint16_t* pb = pls[wave];

  for (int s0 = 0; s0 < 2048; s0 += 32) {
    // keyh B-fragments for 2 key tiles (k-dim = head dim, contiguous in memory)
    const uint16_t* kr0 = kb + (size_t)(s0 + l15) * 256 + khalf * 16;
    const uint16_t* kr1 = kb + (size_t)(s0 + 16 + l15) * 256 + khalf * 16;
    Frag16 bk0, bk1;
    bk0.q[0] = *(const u32x4*)(kr0); bk0.q[1] = *(const u32x4*)(kr0 + 8);
    bk1.q[0] = *(const u32x4*)(kr1); bk1.q[1] = *(const u32x4*)(kr1 + 8);

    v8f sc0 = WMMA_BF16(aq.v, bk0.v, z8);
    v8f sc1 = WMMA_BF16(aq.v, bk1.v, z8);

    // online softmax; rows live in half-wave lane groups (width-16 shuffles)
#pragma unroll
    for (int r = 0; r < 8; ++r) {
      float v0 = sc0[r] * scale, v1 = sc1[r] * scale;
      float tm = fmaxf(v0, v1);
      tm = fmaxf(tm, __shfl_xor(tm, 8, 16));
      tm = fmaxf(tm, __shfl_xor(tm, 4, 16));
      tm = fmaxf(tm, __shfl_xor(tm, 2, 16));
      tm = fmaxf(tm, __shfl_xor(tm, 1, 16));
      float mn = fmaxf(mrow[r], tm);
      float alpha = __expf(mrow[r] - mn);
      float p0 = __expf(v0 - mn), p1 = __expf(v1 - mn);
      float rs = p0 + p1;
      rs += __shfl_xor(rs, 8, 16);
      rs += __shfl_xor(rs, 4, 16);
      rs += __shfl_xor(rs, 2, 16);
      rs += __shfl_xor(rs, 1, 16);
      lrow[r] = lrow[r] * alpha + rs;
      mrow[r] = mn;
      acc0[r] *= alpha; acc1[r] *= alpha;
      int m = r + khalf * 8;                 // C-layout row for this lane
      pb[m * 32 + l15]      = f2bf(p0);
      pb[m * 32 + 16 + l15] = f2bf(p1);
    }

    // P (16x32) as A-fragment from LDS (same-wave DS ops are in-order)
    Frag16 ap;
    ap.q[0] = *(const u32x4*)(pb + l15 * 32 + khalf * 16);
    ap.q[1] = *(const u32x4*)(pb + l15 * 32 + khalf * 16 + 8);

    // V B-fragments (k-dim = key index, contiguous thanks to vT)
    const uint16_t* vr0 = vb + (size_t)l15 * 2048 + s0 + khalf * 16;
    const uint16_t* vr1 = vb + (size_t)(16 + l15) * 2048 + s0 + khalf * 16;
    Frag16 bv0, bv1;
    bv0.q[0] = *(const u32x4*)(vr0); bv0.q[1] = *(const u32x4*)(vr0 + 8);
    bv1.q[0] = *(const u32x4*)(vr1); bv1.q[1] = *(const u32x4*)(vr1 + 8);

    acc0 = WMMA_BF16(ap.v, bv0.v, acc0);
    acc1 = WMMA_BF16(ap.v, bv1.v, acc1);
  }

  uint16_t* aor = AO + ((size_t)(b * 2048 + q0)) * 256 + h * 32;
#pragma unroll
  for (int r = 0; r < 8; ++r) {
    int m = r + khalf * 8;
    float inv = 1.0f / lrow[r];
    aor[(size_t)m * 256 + l15]      = f2bf(acc0[r] * inv);
    aor[(size_t)m * 256 + 16 + l15] = f2bf(acc1[r] * inv);
  }
}

// ---------- launch ----------
extern "C" void kernel_launch(void* const* d_in, const int* in_sizes, int n_in,
                              void* d_out, int out_size, void* d_ws, size_t ws_size,
                              hipStream_t stream) {
  (void)in_sizes; (void)n_in; (void)out_size; (void)ws_size;
  const float* x      = (const float*)d_in[0];
  const float* ln1_g  = (const float*)d_in[1];
  const float* ln1_b  = (const float*)d_in[2];
  const float* w1_W   = (const float*)d_in[3];
  const float* w1_b   = (const float*)d_in[4];
  const float* w2_W   = (const float*)d_in[5];
  const float* w2_b   = (const float*)d_in[6];
  const float* w3_W   = (const float*)d_in[7];
  const float* w3_b   = (const float*)d_in[8];
  const float* q_W    = (const float*)d_in[9];
  const float* q_b    = (const float*)d_in[10];
  const float* k_W    = (const float*)d_in[11];
  const float* k_b    = (const float*)d_in[12];
  const float* v_W    = (const float*)d_in[13];
  const float* v_b    = (const float*)d_in[14];
  const float* o_W    = (const float*)d_in[15];
  const float* o_b    = (const float*)d_in[16];
  const float* gate_W = (const float*)d_in[17];
  const float* gate_b = (const float*)d_in[18];
  const float* up_W   = (const float*)d_in[19];
  const float* up_b   = (const float*)d_in[20];
  const float* down_W = (const float*)d_in[21];
  const float* down_b = (const float*)d_in[22];
  const float* ln2_g  = (const float*)d_in[23];
  const float* ln2_b  = (const float*)d_in[24];
  float* out = (float*)d_out;

  const int M = 4096;                        // B*S rows
  uint8_t* ws = (uint8_t*)d_ws;
  size_t off = 0;
  auto take = [&](size_t bytes) -> void* {
    void* p = ws + off;
    off += (bytes + 255) & ~(size_t)255;
    return p;
  };

  // packed weights (bf16, WMMA-B layout) : bytes = K*N*2
  uint32_t* pw1   = (uint32_t*)take((size_t)1024 * 1024 * 2);
  uint32_t* pw2   = (uint32_t*)take((size_t)1024 * 1024 * 2);
  uint32_t* pw3   = (uint32_t*)take((size_t)1024 * 1024 * 2);
  uint32_t* pgate = (uint32_t*)take((size_t)1024 * 1024 * 2);
  uint32_t* pup   = (uint32_t*)take((size_t)1024 * 1024 * 2);
  uint32_t* pdown = (uint32_t*)take((size_t)1024 * 1024 * 2);
  uint32_t* pq    = (uint32_t*)take((size_t)1024 * 256 * 2);
  uint32_t* pk    = (uint32_t*)take((size_t)1024 * 256 * 2);
  uint32_t* pv    = (uint32_t*)take((size_t)1024 * 256 * 2);
  uint32_t* po    = (uint32_t*)take((size_t)256 * 1024 * 2);
  // activations
  uint16_t* h_bf  = (uint16_t*)take((size_t)M * 1024 * 2);
  float*    qf    = (float*)take((size_t)M * 1024 * 4);    // later: g, then f
  float*    kf    = (float*)take((size_t)M * 1024 * 4);    // later: u
  uint16_t* v_bf  = (uint16_t*)take((size_t)M * 1024 * 2); // later: fin
  uint16_t* qr_bf = (uint16_t*)take((size_t)M * 1024 * 2); // later: x1
  uint16_t* kr_bf = (uint16_t*)take((size_t)M * 1024 * 2);
  uint16_t* qh_bf = (uint16_t*)take((size_t)M * 256 * 2);
  uint16_t* kh_bf = (uint16_t*)take((size_t)M * 256 * 2);
  uint16_t* vh_bf = (uint16_t*)take((size_t)M * 256 * 2);
  uint16_t* vT_bf = (uint16_t*)take((size_t)M * 256 * 2);
  uint16_t* ao_bf = (uint16_t*)take((size_t)M * 256 * 2);
  float*    gf = qf;
  float*    uf = kf;
  uint16_t* fin_bf = v_bf;
  uint16_t* x1_bf  = qr_bf;
  float*    f_f32  = qf;

  auto packW = [&](const float* W, uint32_t* Wp, int K, int N) {
    int tiles = (K / 32) * (N / 16);
    pack_w_kernel<<<dim3((tiles + 7) / 8), dim3(256), 0, stream>>>(W, Wp, K, N);
  };
  packW(w1_W, pw1, 1024, 1024);
  packW(w2_W, pw2, 1024, 1024);
  packW(w3_W, pw3, 1024, 1024);
  packW(gate_W, pgate, 1024, 1024);
  packW(up_W, pup, 1024, 1024);
  packW(down_W, pdown, 1024, 1024);
  packW(q_W, pq, 1024, 256);
  packW(k_W, pk, 1024, 256);
  packW(v_W, pv, 1024, 256);
  packW(o_W, po, 256, 1024);

  auto gemm = [&](const uint16_t* A, const uint32_t* Wp, const float* bias,
                  const float* resid, float* Cf, uint16_t* Cb, int N, int K) {
    gemm_kernel<<<dim3(N / 64, M / 128), dim3(256), 0, stream>>>(A, Wp, bias, resid, Cf, Cb, N, K);
  };

  ln_bf16_kernel<<<dim3(M), dim3(256), 0, stream>>>(x, ln1_g, ln1_b, h_bf);

  gemm(h_bf, pw1, w1_b, nullptr, qf, nullptr, 1024, 1024);        // q (f32 for rope)
  gemm(h_bf, pw2, w2_b, nullptr, kf, nullptr, 1024, 1024);        // k (f32 for rope)
  gemm(h_bf, pw3, w3_b, nullptr, nullptr, v_bf, 1024, 1024);      // v (bf16)

  rope_kernel<<<dim3(M), dim3(256), 0, stream>>>(qf, qr_bf);
  rope_kernel<<<dim3(M), dim3(256), 0, stream>>>(kf, kr_bf);

  gemm(qr_bf, pq, q_b, nullptr, nullptr, qh_bf, 256, 1024);       // qh = rope(q) @ q_W
  gemm(v_bf,  pk, k_b, nullptr, nullptr, kh_bf, 256, 1024);       // keyh = v @ k_W (ref scramble)
  gemm(kr_bf, pv, v_b, nullptr, nullptr, vh_bf, 256, 1024);       // valh = rope(k) @ v_W

  vtrans_kernel<<<dim3(4096), dim3(256), 0, stream>>>(vh_bf, vT_bf);
  attn_kernel<<<dim3(256), dim3(256), 0, stream>>>(qh_bf, kh_bf, vT_bf, ao_bf);

  gemm(ao_bf, po, o_b, x, nullptr, x1_bf, 1024, 256);             // x1 = x + ao@o_W + o_b

  gemm(x1_bf, pgate, gate_b, nullptr, gf, nullptr, 1024, 1024);
  gemm(x1_bf, pup,   up_b,   nullptr, uf, nullptr, 1024, 1024);
  silu_mul_kernel<<<dim3((M * 1024) / 256), dim3(256), 0, stream>>>(gf, uf, fin_bf, M * 1024);
  gemm(fin_bf, pdown, down_b, nullptr, f_f32, nullptr, 1024, 1024);

  final_kernel<<<dim3(M), dim3(256), 0, stream>>>(f_f32, ln2_g, ln2_b, out);
}
